// Point_Matching_Loss_84396107366436
// MI455X (gfx1250) — compile-verified, matrix operand order checked by reading, not run on hardware
//
#include <hip/hip_runtime.h>
#include <hip/hip_bf16.h>

#define B_    4
#define C_    128
#define H_    240
#define W_    240
#define HW_   (H_*W_)
#define NK_   400
#define CELL_ 12
#define GH_   20
#define EPSF  1e-6f

typedef __attribute__((ext_vector_type(16))) _Float16 v16h;
typedef __attribute__((ext_vector_type(8)))  _Float16 v8h;
typedef __attribute__((ext_vector_type(8)))  float    v8f;

__device__ __forceinline__ float waveMax(float v) {
#pragma unroll
  for (int m = 16; m; m >>= 1) v = fmaxf(v, __shfl_xor(v, m, 32));
  return v;
}
__device__ __forceinline__ float waveSum(float v) {
#pragma unroll
  for (int m = 16; m; m >>= 1) v += __shfl_xor(v, m, 32);
  return v;
}
// sum across the 16 lanes of each half-wave (lanes 0-15 / 16-31 independent)
__device__ __forceinline__ float halfSum(float v) {
#pragma unroll
  for (int m = 1; m <= 8; m <<= 1) v += __shfl_xor(v, m, 32);
  return v;
}

// ---------------------------------------------------------------------------
// Kernel 1: normalize desc2 over W (reference quirk: axis=-1 of (b,c,h,w)),
// transpose to key-major [b][m][c] and convert to f16 for WMMA B operand.
// One wave per (b,c,h) row.
// ---------------------------------------------------------------------------
__global__ __launch_bounds__(32) void k_keys(const float* __restrict__ desc2,
                                             _Float16* __restrict__ keys) {
  int idx = blockIdx.x;
  int h = idx % H_;
  int c = (idx / H_) % C_;
  int b = idx / (H_ * C_);
  int lane = threadIdx.x;
  const float* row = desc2 + (((size_t)b * C_ + c) * H_ + h) * W_;
  float v[8];
  float ssq = 0.f;
#pragma unroll
  for (int k = 0; k < 8; ++k) {
    int w = lane + k * 32;
    float x = (w < W_) ? row[w] : 0.f;
    v[k] = x;
    ssq += x * x;
  }
  ssq = waveSum(ssq);
  float inv = 1.f / (sqrtf(ssq) + EPSF);
  _Float16* out = keys + ((size_t)b * HW_ + (size_t)h * W_) * C_ + c;
#pragma unroll
  for (int k = 0; k < 8; ++k) {
    int w = lane + k * 32;
    if (w < W_) out[(size_t)w * C_] = (_Float16)(v[k] * inv);
  }
}

// ---------------------------------------------------------------------------
// Kernel 2: spatial softmax keypoints.  One wave per (b, keypoint).
// ---------------------------------------------------------------------------
__global__ __launch_bounds__(32) void k_spatial(const float* __restrict__ loc,
                                                float* __restrict__ ps) {
  int id = blockIdx.x;              // 0..B*NK-1
  int b = id / NK_;
  int kp = id % NK_;
  int ghi = kp / GH_;
  int gwi = kp % GH_;
  int lane = threadIdx.x;
  const float* base = loc + (size_t)b * HW_;
  float v[5];
#pragma unroll
  for (int k = 0; k < 5; ++k) {
    int d = lane + k * 32;
    if (d < CELL_ * CELL_) {
      int cy = d / CELL_, cx = d % CELL_;
      v[k] = base[(ghi * CELL_ + cy) * W_ + gwi * CELL_ + cx];
    } else {
      v[k] = -3.0e38f;
    }
  }
  float mx = v[0];
#pragma unroll
  for (int k = 1; k < 5; ++k) mx = fmaxf(mx, v[k]);
  mx = waveMax(mx);
  float se = 0.f, sx = 0.f, sy = 0.f;
#pragma unroll
  for (int k = 0; k < 5; ++k) {
    int d = lane + k * 32;
    if (d < CELL_ * CELL_) {
      float e = __expf(v[k] - mx);
      se += e;
      sx += e * (float)(d % CELL_);
      sy += e * (float)(d / CELL_);
    }
  }
  se = waveSum(se);
  sx = waveSum(sx);
  sy = waveSum(sy);
  if (lane == 0) {
    // reference quirk: x gets the row-grid offset (ghi*cell), y gets gwi*cell
    ps[(size_t)id * 2 + 0] = sx / se + (float)(ghi * CELL_);
    ps[(size_t)id * 2 + 1] = sy / se + (float)(gwi * CELL_);
  }
}

__device__ __forceinline__ float bilin(const float* __restrict__ base, float x,
                                       float y) {
  x = fminf(fmaxf(x, 0.f), (float)(W_ - 1));
  y = fminf(fmaxf(y, 0.f), (float)(H_ - 1));
  float x0f = floorf(x), y0f = floorf(y);
  int x0 = (int)x0f, y0 = (int)y0f;
  int x1 = min(x0 + 1, W_ - 1), y1 = min(y0 + 1, H_ - 1);
  float wx = x - x0f, wy = y - y0f;
  float v00 = base[y0 * W_ + x0], v01 = base[y0 * W_ + x1];
  float v10 = base[y1 * W_ + x0], v11 = base[y1 * W_ + x1];
  return v00 * (1.f - wx) * (1.f - wy) + v01 * wx * (1.f - wy) +
         v10 * (1.f - wx) * wy + v11 * wx * wy;
}

// ---------------------------------------------------------------------------
// Kernel 3: ds = l2norm_C(bilinear(desc1, ps)), f16 copy for WMMA A,
// ss = bilinear(sc1, ps).  One 128-thread block per (b, keypoint).
// ---------------------------------------------------------------------------
__global__ __launch_bounds__(128) void k_sample1(
    const float* __restrict__ desc1, const float* __restrict__ sc1,
    const float* __restrict__ ps, float* __restrict__ ds,
    _Float16* __restrict__ q16, float* __restrict__ ss_out) {
  __shared__ float sm[4];
  int id = blockIdx.x;
  int b = id / NK_;
  int c = threadIdx.x;
  float x = ps[(size_t)id * 2], y = ps[(size_t)id * 2 + 1];
  const float* base = desc1 + ((size_t)b * C_ + c) * HW_;
  float v = bilin(base, x, y);
  float sq = v * v;
#pragma unroll
  for (int m = 16; m; m >>= 1) sq += __shfl_xor(sq, m, 32);
  int wave = threadIdx.x >> 5;
  if ((threadIdx.x & 31) == 0) sm[wave] = sq;
  __syncthreads();
  float nrm = sqrtf(sm[0] + sm[1] + sm[2] + sm[3]);
  float dn = v / (nrm + EPSF);
  ds[(size_t)id * C_ + c] = dn;
  q16[(size_t)id * C_ + c] = (_Float16)dn;
  if (threadIdx.x == 0) ss_out[id] = bilin(sc1 + (size_t)b * HW_, x, y);
}

// ---------------------------------------------------------------------------
// Kernel 4: fused attention  pd = softmax(ds · keys^T) · coords
// One workgroup (8 waves) per (batch, 16-query tile); waves split the 3600 key
// tiles (450 each).  Scores via chained v_wmma_f32_16x16x32_f16, explicit
// double-buffered B operand (loads for tile k+1 in flight during WMMAs of
// tile k).  Softmax without running max: |score| <= sqrt(128) by
// Cauchy-Schwarz, so exp() cannot overflow f32; per-lane plain sums, merged
// by shfl butterfly + LDS across waves.
// ---------------------------------------------------------------------------
__device__ __forceinline__ void loadB(const _Float16* __restrict__ krow,
                                      int hsel, v16h (&Bm)[4]) {
#pragma unroll
  for (int c = 0; c < 4; ++c) {
    int k0 = c * 32 + hsel * 8;
    v8h lo = *reinterpret_cast<const v8h*>(krow + k0);
    v8h hi = *reinterpret_cast<const v8h*>(krow + k0 + 16);
    Bm[c] = __builtin_shufflevector(lo, hi, 0, 1, 2, 3, 4, 5, 6, 7, 8, 9, 10,
                                    11, 12, 13, 14, 15);
  }
}

__global__ __launch_bounds__(256) void k_attn(const _Float16* __restrict__ q16,
                                              const _Float16* __restrict__ keys,
                                              float* __restrict__ pd) {
  __shared__ float Ls[8][16], Li[8][16], Lj[8][16];
  int wg = blockIdx.x;
  int b = wg / 25, qt = wg % 25;
  int lane = threadIdx.x & 31;
  int wave = threadIdx.x >> 5;
  int hsel = lane >> 4;   // 0: K-chunk {0..7,16..23}, 1: {8..15,24..31}
  int l16 = lane & 15;

  // A operand: query rows, ISA 16-bit A-matrix lane/K mapping
  const _Float16* qrow = q16 + ((size_t)b * NK_ + qt * 16 + l16) * C_;
  v16h A[4];
#pragma unroll
  for (int c = 0; c < 4; ++c) {
    int k0 = c * 32 + hsel * 8;
    v8h lo = *reinterpret_cast<const v8h*>(qrow + k0);
    v8h hi = *reinterpret_cast<const v8h*>(qrow + k0 + 16);
    A[c] = __builtin_shufflevector(lo, hi, 0, 1, 2, 3, 4, 5, 6, 7, 8, 9, 10,
                                   11, 12, 13, 14, 15);
  }

  float s_[8], si_[8], sj_[8];
#pragma unroll
  for (int r = 0; r < 8; ++r) { s_[r] = 0.f; si_[r] = 0.f; sj_[r] = 0.f; }

  // this lane's key index starts at wave*16 + l16 (< 240), advances by 128
  float fi = 0.f;
  float fj = (float)(wave * 16 + l16);
  const size_t tileStep = (size_t)128 * C_;  // 8 tiles of 16 rows
  const _Float16* krow = keys + ((size_t)b * HW_ + wave * 16 + l16) * C_;

  v16h B0[4], B1[4];
  loadB(krow, hsel, B0);

  for (int k = 0; k < 450; k += 2) {
    // ---- stage 0: consume B0 (tile k), fetch B1 (tile k+1) ----
    const _Float16* k1 = krow + tileStep;
    loadB(k1, hsel, B1);
    __builtin_prefetch(k1 + tileStep, 0, 1);
    v8f acc = {};
#pragma unroll
    for (int c = 0; c < 4; ++c)
      acc = __builtin_amdgcn_wmma_f32_16x16x32_f16(false, A[c], false, B0[c],
                                                   (short)0, acc, false, false);
#pragma unroll
    for (int r = 0; r < 8; ++r) {
      float p = __expf(acc[r]);
      s_[r] += p;
      si_[r] = fmaf(p, fi, si_[r]);
      sj_[r] = fmaf(p, fj, sj_[r]);
    }
    fj += 128.f;
    if (fj >= 240.f) { fj -= 240.f; fi += 1.f; }

    // ---- stage 1: consume B1 (tile k+1), fetch B0 (tile k+2) ----
    const _Float16* k2 = k1 + tileStep;
    const _Float16* k2c = (k + 2 < 450) ? k2 : krow;  // clamp past-end fetch
    loadB(k2c, hsel, B0);
    __builtin_prefetch(k2c + tileStep, 0, 1);
    v8f acc2 = {};
#pragma unroll
    for (int c = 0; c < 4; ++c)
      acc2 = __builtin_amdgcn_wmma_f32_16x16x32_f16(false, A[c], false, B1[c],
                                                    (short)0, acc2, false, false);
#pragma unroll
    for (int r = 0; r < 8; ++r) {
      float p = __expf(acc2[r]);
      s_[r] += p;
      si_[r] = fmaf(p, fi, si_[r]);
      sj_[r] = fmaf(p, fj, sj_[r]);
    }
    fj += 128.f;
    if (fj >= 240.f) { fj -= 240.f; fi += 1.f; }

    krow = k2;
  }

  // butterfly sum across the 16 lanes of each half-wave
#pragma unroll
  for (int r = 0; r < 8; ++r) {
    s_[r] = halfSum(s_[r]);
    si_[r] = halfSum(si_[r]);
    sj_[r] = halfSum(sj_[r]);
  }
  if (l16 == 0) {
#pragma unroll
    for (int r = 0; r < 8; ++r) {
      int M = r + 8 * hsel;
      Ls[wave][M] = s_[r];
      Li[wave][M] = si_[r];
      Lj[wave][M] = sj_[r];
    }
  }
  __syncthreads();
  if (threadIdx.x < 16) {
    int M = threadIdx.x;
    float ssv = 0.f, ii = 0.f, jj = 0.f;
#pragma unroll
    for (int w = 0; w < 8; ++w) {
      ssv += Ls[w][M];
      ii += Li[w][M];
      jj += Lj[w][M];
    }
    size_t q = (size_t)b * NK_ + (size_t)qt * 16 + M;
    pd[q * 2 + 0] = ii / ssv;
    pd[q * 2 + 1] = jj / ssv;
  }
}

// ---------------------------------------------------------------------------
// Kernel 5: dd = l2norm_C(bilinear(desc2, pd)); sd = bilinear(sc2, pd);
// w_ = ((ds·dd)+1) * ss*sd / 2.
// ---------------------------------------------------------------------------
__global__ __launch_bounds__(128) void k_sample2(
    const float* __restrict__ desc2, const float* __restrict__ sc2,
    const float* __restrict__ pd, const float* __restrict__ ds,
    const float* __restrict__ ss_in, float* __restrict__ w_out) {
  __shared__ float sm[4];
  int id = blockIdx.x;
  int b = id / NK_;
  int c = threadIdx.x;
  float x = pd[(size_t)id * 2], y = pd[(size_t)id * 2 + 1];
  const float* base = desc2 + ((size_t)b * C_ + c) * HW_;
  float v = bilin(base, x, y);
  float sq = v * v;
#pragma unroll
  for (int m = 16; m; m >>= 1) sq += __shfl_xor(sq, m, 32);
  int wave = threadIdx.x >> 5;
  if ((threadIdx.x & 31) == 0) sm[wave] = sq;
  __syncthreads();
  float nrm = sqrtf(sm[0] + sm[1] + sm[2] + sm[3]);
  __syncthreads();
  float dn = v / (nrm + EPSF);
  float dot = dn * ds[(size_t)id * C_ + c];
#pragma unroll
  for (int m = 16; m; m >>= 1) dot += __shfl_xor(dot, m, 32);
  if ((threadIdx.x & 31) == 0) sm[wave] = dot;
  __syncthreads();
  if (threadIdx.x == 0) {
    float d4 = sm[0] + sm[1] + sm[2] + sm[3];
    float sd = bilin(sc2 + (size_t)b * HW_, x, y);
    w_out[id] = (d4 + 1.f) * (ss_in[id] * sd) * 0.5f;
  }
}

// ---------------------------------------------------------------------------
// Kernel 6: weighted 2x2 Procrustes (closed-form Kabsch) + loss.
// One wave per batch; shuffle-reduce 9 sufficient statistics.
// ---------------------------------------------------------------------------
__global__ __launch_bounds__(128) void k_pose(const float* __restrict__ ps,
                                              const float* __restrict__ pd,
                                              const float* __restrict__ w_,
                                              const float* __restrict__ P,
                                              float* __restrict__ out) {
  __shared__ float Lt[4], LR[4];
  int b = threadIdx.x >> 5;
  int lane = threadIdx.x & 31;
  float Sw = 0, Sqsx = 0, Sqsy = 0, Sqdx = 0, Sqdy = 0;
  float Sxx = 0, Sxy = 0, Syx = 0, Syy = 0;
  for (int n = lane; n < NK_; n += 32) {
    size_t i = (size_t)b * NK_ + n;
    float w = w_[i];
    float qsx = (ps[i * 2] - 119.5f) * 0.25f;
    float qsy = (119.5f - ps[i * 2 + 1]) * 0.25f;
    float qdx = (pd[i * 2] - 119.5f) * 0.25f;
    float qdy = (119.5f - pd[i * 2 + 1]) * 0.25f;
    Sw += w;
    Sqsx += w * qsx; Sqsy += w * qsy; Sqdx += w * qdx; Sqdy += w * qdy;
    Sxx += w * qsx * qdx; Sxy += w * qsx * qdy;
    Syx += w * qsy * qdx; Syy += w * qsy * qdy;
  }
  Sw = waveSum(Sw);
  Sqsx = waveSum(Sqsx); Sqsy = waveSum(Sqsy);
  Sqdx = waveSum(Sqdx); Sqdy = waveSum(Sqdy);
  Sxx = waveSum(Sxx); Sxy = waveSum(Sxy);
  Syx = waveSum(Syx); Syy = waveSum(Syy);
  if (lane == 0) {
    float wsum = Sw + EPSF;
    float asx = Sqsx / wsum, asy = Sqsy / wsum;
    float adx = Sqdx / wsum, ady = Sqdy / wsum;
    float S00 = Sxx - asx * Sqdx - Sqsx * adx + Sw * asx * adx;
    float S01 = Sxy - asx * Sqdy - Sqsx * ady + Sw * asx * ady;
    float S10 = Syx - asy * Sqdx - Sqsy * adx + Sw * asy * adx;
    float S11 = Syy - asy * Sqdy - Sqsy * ady + Sw * asy * ady;
    // R = V d U^T (SVD) == proper rotation maximizing tr(R S):
    float c0 = S00 + S11, s0 = S01 - S10;
    float rr = sqrtf(c0 * c0 + s0 * s0) + 1e-30f;
    float Rc = c0 / rr, Rs = s0 / rr;
    float t0 = adx - (Rc * asx - Rs * asy);
    float t1 = ady - (Rs * asx + Rc * asy);
    float tr0 = P[b * 9 + 2], tr1 = P[b * 9 + 5];
    float lt = sqrtf((tr0 - t0) * (tr0 - t0) + (tr1 - t1) * (tr1 - t1));
    float Rr00 = P[b * 9 + 0], Rr01 = P[b * 9 + 1];
    float Rr10 = P[b * 9 + 3], Rr11 = P[b * 9 + 4];
    float M00 = Rr00 * Rc - Rr01 * Rs - 1.f;
    float M01 = Rr00 * Rs + Rr01 * Rc;
    float M10 = Rr10 * Rc - Rr11 * Rs;
    float M11 = Rr10 * Rs + Rr11 * Rc - 1.f;
    Lt[b] = lt;
    LR[b] = sqrtf(M00 * M00 + M01 * M01 + M10 * M10 + M11 * M11);
  }
  __syncthreads();
  if (threadIdx.x == 0) {
    float lt = 0.25f * (Lt[0] + Lt[1] + Lt[2] + Lt[3]);
    float lR = 0.25f * (LR[0] + LR[1] + LR[2] + LR[3]);
    out[0] = lt + 10.f * lR;
  }
}

// ---------------------------------------------------------------------------
extern "C" void kernel_launch(void* const* d_in, const int* in_sizes, int n_in,
                              void* d_out, int out_size, void* d_ws,
                              size_t ws_size, hipStream_t stream) {
  (void)in_sizes; (void)n_in; (void)out_size; (void)ws_size;
  const float* loc1 = (const float*)d_in[0];
  const float* sc1 = (const float*)d_in[1];
  const float* desc1 = (const float*)d_in[2];
  const float* sc2 = (const float*)d_in[3];
  const float* desc2 = (const float*)d_in[4];
  const float* ptrans = (const float*)d_in[5];
  float* outp = (float*)d_out;

  char* ws = (char*)d_ws;
  size_t off = 0;
  _Float16* keys = (_Float16*)(ws + off); off += (size_t)B_ * HW_ * C_ * 2;  // 58.98 MB
  _Float16* q16 = (_Float16*)(ws + off);  off += (size_t)B_ * NK_ * C_ * 2;
  float* ds = (float*)(ws + off);         off += (size_t)B_ * NK_ * C_ * 4;
  float* ps = (float*)(ws + off);         off += (size_t)B_ * NK_ * 2 * 4;
  float* pd = (float*)(ws + off);         off += (size_t)B_ * NK_ * 2 * 4;
  float* ss = (float*)(ws + off);         off += (size_t)B_ * NK_ * 4;
  float* wv = (float*)(ws + off);         off += (size_t)B_ * NK_ * 4;
  (void)off;

  k_keys<<<dim3(B_ * C_ * H_), dim3(32), 0, stream>>>(desc2, keys);
  k_spatial<<<dim3(B_ * NK_), dim3(32), 0, stream>>>(loc1, ps);
  k_sample1<<<dim3(B_ * NK_), dim3(128), 0, stream>>>(desc1, sc1, ps, ds, q16, ss);
  k_attn<<<dim3(B_ * 25), dim3(256), 0, stream>>>(q16, keys, pd);
  k_sample2<<<dim3(B_ * NK_), dim3(128), 0, stream>>>(desc2, sc2, pd, ds, ss, wv);
  k_pose<<<dim3(1), dim3(128), 0, stream>>>(ps, pd, wv, ptrans, outp);
}
